// CausalSelfAttention_4080218931351
// MI455X (gfx1250) — matrix operand order repaired and checked by reading, not compiled, not run on hardware
//
#include <hip/hip_runtime.h>
#include <hip/hip_bf16.h>

// ---------------------------------------------------------------------------
// Causal self-attention, f16 WMMA + Tensor-Data-Mover pipeline for gfx1250.
// B=2, S=2048, H=2048, NH=16, HD=128.
//  - all matmuls on v_wmma_f32_16x16x32_f16 (f32 accumulate)
//  - tile movement via TDM (tensor_load_to_lds + s_wait_tensorcnt)
//  - GEMMs double-buffer LDS so the TDM DMA overlaps WMMA compute
// ---------------------------------------------------------------------------

#define B_  2
#define S_  2048
#define H_  2048
#define NH_ 16
#define HD_ 128

typedef __attribute__((ext_vector_type(16))) _Float16 v16h;
typedef __attribute__((ext_vector_type(8)))  _Float16 v8h;
typedef __attribute__((ext_vector_type(8)))  float    v8f;
typedef unsigned int u32x4 __attribute__((ext_vector_type(4)));
typedef int          i32x4 __attribute__((ext_vector_type(4)));
typedef int          i32x8 __attribute__((ext_vector_type(8)));

#if defined(__HIP_DEVICE_COMPILE__) && __has_builtin(__builtin_amdgcn_tensor_load_to_lds) && \
    __has_builtin(__builtin_amdgcn_s_wait_tensorcnt)
#define USE_TDM 1
#else
#define USE_TDM 0
#endif

// D = A(16x32 f16) * B(32x16 f16) + C(16x16 f32)
static __device__ __forceinline__ v8f wmma16(v16h a, v16h b, v8f c) {
  return __builtin_amdgcn_wmma_f32_16x16x32_f16(
      /*neg_a=*/false, a, /*neg_b=*/false, b,
      /*c_mod=*/(short)0, c, /*reuse_a=*/false, /*reuse_b=*/false);
}

#if USE_TDM
// TDM 2-D tile load: rows x width f16 elements, row stride stride_elems
// (f16 units), into LDS at byte offset lds_off, appending an 8-half
// (4-DWORD) pad per row.  pad_icode: 2^(code+1) DWORDs between pads.
// D# layout per CDNA5 ISA ch.8.
static __device__ __forceinline__ void tdm_load_2d(unsigned lds_off, const void* gptr,
                                                   unsigned width_elems, unsigned rows,
                                                   unsigned stride_elems, unsigned pad_icode) {
  const unsigned long long ga = (unsigned long long)(size_t)gptr;
  u32x4 g0;
  g0[0] = 1u;                                             // count=1 (valid user D#)
  g0[1] = lds_off;                                        // lds_addr (bytes)
  g0[2] = (unsigned)(ga & 0xFFFFFFFFu);                   // global_addr[31:0]
  g0[3] = (unsigned)((ga >> 32) & 0x01FFFFFFu)            // global_addr[56:32]
          | (2u << 30);                                   // type=2 ("image")
  i32x8 g1;
  g1[0] = (int)((1u << 16)                                // data_size=1 -> 2 bytes
                | (1u << 20)                              // pad_enable
                | (pad_icode << 22)                       // pad_interval
                | (3u << 25));                            // pad_amount: 4 DWORDs
  g1[1] = (int)((width_elems & 0xFFFFu) << 16);           // tensor_dim0[15:0]
  g1[2] = (int)(((width_elems >> 16) & 0xFFFFu) | ((rows & 0xFFFFu) << 16));
  g1[3] = (int)(((rows >> 16) & 0xFFFFu) | ((width_elems & 0xFFFFu) << 16)); // tile_dim0
  g1[4] = (int)(rows & 0xFFFFu);                          // tile_dim1 (tile_dim2=0)
  g1[5] = (int)stride_elems;                              // tensor_dim0_stride[31:0]
  g1[6] = 0;
  g1[7] = 0;
  const i32x4 z4 = {0, 0, 0, 0};
#if __clang_major__ >= 23
  const i32x8 z8 = {0, 0, 0, 0, 0, 0, 0, 0};
  __builtin_amdgcn_tensor_load_to_lds(g0, g1, z4, z4, z8, 0);
#else
  __builtin_amdgcn_tensor_load_to_lds(g0, g1, z4, z4, 0);
#endif
}
#endif  // USE_TDM

// A-fragment (16x32, MxK), source row-major with leading dim ld (halfs).
static __device__ __forceinline__ v16h load_a_frag(const _Float16* __restrict__ base, int ld) {
  const int lane = threadIdx.x & 31;
  const int hf = lane >> 4;
  const int m  = lane & 15;
  v16h a;
#pragma unroll
  for (int e = 0; e < 16; ++e) {
    const int v = e >> 1, p = e & 1;
    const int k = hf * 16 + 2 * v + p;
    a[e] = base[m * ld + k];
  }
  return a;
}

// B-fragment (32x16, KxN), source row-major [K x N] with leading dim ld.
static __device__ __forceinline__ v16h load_b_frag(const _Float16* __restrict__ base, int ld) {
  const int lane = threadIdx.x & 31;
  const int n  = lane & 15;
  const int kb = (lane >> 4) * 16;
  v16h b;
#pragma unroll
  for (int e = 0; e < 16; ++e) b[e] = base[(kb + e) * ld + n];
  return b;
}

// B-fragment where the source holds B transposed: row-major [N x K].
static __device__ __forceinline__ v16h load_bt_frag(const _Float16* __restrict__ base, int ld) {
  const int lane = threadIdx.x & 31;
  const int n  = lane & 15;
  const int kb = (lane >> 4) * 16;
  v16h b;
#pragma unroll
  for (int e = 0; e < 16; ++e) b[e] = base[n * ld + kb + e];
  return b;
}

// ---------------------------------------------------------------------------
__global__ __launch_bounds__(256) void cast_f16_kernel(const float* __restrict__ src,
                                                       _Float16* __restrict__ dst, int n) {
  int i = blockIdx.x * blockDim.x + threadIdx.x;
  const int stride = gridDim.x * blockDim.x;
  for (; i < n; i += stride) dst[i] = (_Float16)src[i];
}

// ---------------------------------------------------------------------------
// C[M,N] = A[M,K] (f16, row-major) @ W[K,N] (f16, row-major) + bias[N] (f32).
// Block tile 128x128, 8 waves; wave tile 32x64 (2x4 of 16x16).
// Double-buffered LDS, TDM prefetch of tile i+1 overlaps compute of tile i.
// Per-buffer carve (halfs): As 128x40 = 5120, Bs 32x136 = 4352 -> 9472 halfs
// (18944 B); two buffers = 37888 B.
// ---------------------------------------------------------------------------
template <bool OUT_F16>
__global__ __launch_bounds__(256) void gemm128(const _Float16* __restrict__ A,
                                               const _Float16* __restrict__ W,
                                               const float* __restrict__ bias,
                                               void* __restrict__ Cout,
                                               int M, int N, int K) {
  constexpr int ASZ = 128 * 40;          // halfs
  constexpr int BSZ = 32 * 136;          // halfs
  constexpr int BUFH = ASZ + BSZ;        // 9472 halfs per buffer
  constexpr unsigned BUFB = BUFH * 2;    // 18944 bytes per buffer
  __shared__ __align__(16) _Float16 gsmem[2 * BUFH];

  const int tid  = threadIdx.x;
  const int lane = tid & 31;
  const int w    = tid >> 5;
  const int hf   = lane >> 4;
  const int nl   = lane & 15;
  const int m0 = blockIdx.y * 128;
  const int n0 = blockIdx.x * 128;
  const int wm = (w >> 1) * 32;
  const int wn = (w & 1) * 64;

  v8f acc[2][4];
  const v8f vzero = {};
#pragma unroll
  for (int mt = 0; mt < 2; ++mt)
#pragma unroll
    for (int nt = 0; nt < 4; ++nt) acc[mt][nt] = vzero;

#if USE_TDM
  // Prologue: load tile 0 into buffer 0.
  if (w == 0) {
    tdm_load_2d(0u, &A[m0 * K], 32u, 128u, (unsigned)K, 3u);
    tdm_load_2d((unsigned)(ASZ * 2), &W[n0], 128u, 32u, (unsigned)N, 5u);
    __builtin_amdgcn_s_wait_tensorcnt(0);
  }
  __syncthreads();
  int nbuf = 0;
  for (int k0 = 0; k0 < K; k0 += 32) {
    // Prefetch tile i+1 into the other buffer; DMA overlaps the WMMAs below.
    if (w == 0 && (k0 + 32) < K) {
      const unsigned boff = (unsigned)(nbuf ^ 1) * BUFB;
      tdm_load_2d(boff, &A[m0 * K + k0 + 32], 32u, 128u, (unsigned)K, 3u);
      tdm_load_2d(boff + (unsigned)(ASZ * 2), &W[(k0 + 32) * N + n0], 128u, 32u,
                  (unsigned)N, 5u);
    }
    const _Float16* As = gsmem + nbuf * BUFH;
    const _Float16* Bs = As + ASZ;

    v16h af[2];
#pragma unroll
    for (int mt = 0; mt < 2; ++mt) af[mt] = load_a_frag(&As[(wm + mt * 16) * 40], 40);
#pragma unroll
    for (int nt = 0; nt < 4; ++nt) {
      const v16h bf = load_b_frag(&Bs[wn + nt * 16], 136);
#pragma unroll
      for (int mt = 0; mt < 2; ++mt) acc[mt][nt] = wmma16(af[mt], bf, acc[mt][nt]);
    }

    if (w == 0) __builtin_amdgcn_s_wait_tensorcnt(0);  // drain prefetch
    __syncthreads();
    nbuf ^= 1;
  }
#else
  _Float16* As = gsmem;
  _Float16* Bs = gsmem + ASZ;
  for (int k0 = 0; k0 < K; k0 += 32) {
    __syncthreads();
#pragma unroll
    for (int j = 0; j < 2; ++j) {
      const int c = tid + j * 256;
      const int row = c >> 2, col = (c & 3) << 3;
      *(v8h*)&As[row * 40 + col] = *(const v8h*)&A[(m0 + row) * K + k0 + col];
    }
#pragma unroll
    for (int j = 0; j < 2; ++j) {
      const int c = tid + j * 256;
      const int row = c >> 4, col = (c & 15) << 3;
      *(v8h*)&Bs[row * 136 + col] = *(const v8h*)&W[(k0 + row) * N + n0 + col];
    }
    __syncthreads();

    v16h af[2];
#pragma unroll
    for (int mt = 0; mt < 2; ++mt) af[mt] = load_a_frag(&As[(wm + mt * 16) * 40], 40);
#pragma unroll
    for (int nt = 0; nt < 4; ++nt) {
      const v16h bf = load_b_frag(&Bs[wn + nt * 16], 136);
#pragma unroll
      for (int mt = 0; mt < 2; ++mt) acc[mt][nt] = wmma16(af[mt], bf, acc[mt][nt]);
    }
  }
#endif

#pragma unroll
  for (int nt = 0; nt < 4; ++nt) {
    const float bv = bias[n0 + wn + nt * 16 + nl];
#pragma unroll
    for (int mt = 0; mt < 2; ++mt) {
#pragma unroll
      for (int r = 0; r < 8; ++r) {
        const int row = m0 + wm + mt * 16 + r + 8 * hf;
        const int col = n0 + wn + nt * 16 + nl;
        const float v = acc[mt][nt][r] + bv;
        if (OUT_F16) ((_Float16*)Cout)[row * N + col] = (_Float16)v;
        else         ((float*)Cout)[row * N + col] = v;
      }
    }
  }
}

// ---------------------------------------------------------------------------
// Flash attention: one WG per (q-block of 128 rows, head, batch).
// LDS carve (byte offsets):  KL @ 0     : 64 x 136 halfs
//                            VL @ 17408 : 64 x 136 halfs
//                            PL @ 34816 : 128 x 72 halfs (wave-private rows)
//                            QS @ 0     : 128 x 136 staging (pre-loop only)
// ---------------------------------------------------------------------------
__global__ __launch_bounds__(256) void attn128(const _Float16* __restrict__ Qg,
                                               const _Float16* __restrict__ Kg,
                                               const _Float16* __restrict__ Vg,
                                               _Float16* __restrict__ Cg) {
  __shared__ __align__(16) _Float16 smem[26624];  // 52 KB
  _Float16* KL = smem;
  _Float16* VL = smem + 64 * 136;
  _Float16* PL = smem + 2 * 64 * 136;
  _Float16* QS = smem;

  const int tid  = threadIdx.x;
  const int lane = tid & 31;
  const int w    = tid >> 5;
  const int hf   = lane >> 4;
  const int nl   = lane & 15;
  const int qb = blockIdx.x, h = blockIdx.y, b = blockIdx.z;
  const int m0 = qb * 128;
  const int rowbase = b * S_ + m0;
  const int colbase = h * HD_;
  const float scale = 0.08838834764831845f;  // HD^-0.5

  // ---- Stage Q block (128 x 128) through LDS, build A-fragments in regs ----
#if USE_TDM
  if (w == 0) {
    tdm_load_2d(0u, &Qg[rowbase * H_ + colbase], 128u, 128u, (unsigned)H_, 5u);
    __builtin_amdgcn_s_wait_tensorcnt(0);
  }
#else
#pragma unroll
  for (int j = 0; j < 8; ++j) {
    const int c = tid + j * 256;
    const int row = c >> 4, col = (c & 15) << 3;
    *(v8h*)&QS[row * 136 + col] = *(const v8h*)&Qg[(rowbase + row) * H_ + colbase + col];
  }
#endif
  __syncthreads();
  v16h qf[4];
#pragma unroll
  for (int ks = 0; ks < 4; ++ks) qf[ks] = load_a_frag(&QS[(w * 16) * 136 + ks * 32], 136);

  const v8f vzero = {};
  v8f oacc[8];
#pragma unroll
  for (int t = 0; t < 8; ++t) oacc[t] = vzero;
  float row_m[8], row_l[8];
#pragma unroll
  for (int r = 0; r < 8; ++r) { row_m[r] = -1e30f; row_l[r] = 0.0f; }

  const int kbmax = 2 * qb + 1;  // causal: keys < (qb+1)*128
  for (int kb = 0; kb <= kbmax; ++kb) {
    __syncthreads();
#if USE_TDM
    if (w == 0) {
      const int g = (b * S_ + kb * 64) * H_ + colbase;
      tdm_load_2d(0u,     &Kg[g], 128u, 64u, (unsigned)H_, 5u);
      tdm_load_2d(17408u, &Vg[g], 128u, 64u, (unsigned)H_, 5u);
      __builtin_amdgcn_s_wait_tensorcnt(0);
    }
#else
#pragma unroll
    for (int j = 0; j < 4; ++j) {
      const int c = tid + j * 256;
      const int row = c >> 4, col = (c & 15) << 3;
      const int g = (b * S_ + kb * 64 + row) * H_ + colbase + col;
      *(v8h*)&KL[row * 136 + col] = *(const v8h*)&Kg[g];
      *(v8h*)&VL[row * 136 + col] = *(const v8h*)&Vg[g];
    }
#endif
    __syncthreads();

    // ---- scores S = Q * K^T (wave: 16 M-rows x 64 keys) ----
    v8f sc[4];
#pragma unroll
    for (int nt = 0; nt < 4; ++nt) sc[nt] = vzero;
#pragma unroll
    for (int ks = 0; ks < 4; ++ks)
#pragma unroll
      for (int nt = 0; nt < 4; ++nt) {
        const v16h bf = load_bt_frag(&KL[(nt * 16) * 136 + ks * 32], 136);
        sc[nt] = wmma16(qf[ks], bf, sc[nt]);
      }

    // ---- scale + causal mask ----
#pragma unroll
    for (int nt = 0; nt < 4; ++nt)
#pragma unroll
      for (int r = 0; r < 8; ++r) {
        const int ng = kb * 64 + nt * 16 + nl;
        const int mg = m0 + w * 16 + r + 8 * hf;
        const float v = sc[nt][r] * scale;
        sc[nt][r] = (ng > mg) ? -1e30f : v;
      }

    // ---- row max (reduce over 16 lanes in half-wave) ----
    float tm[8];
#pragma unroll
    for (int r = 0; r < 8; ++r) {
      float m = sc[0][r];
#pragma unroll
      for (int nt = 1; nt < 4; ++nt) m = fmaxf(m, sc[nt][r]);
      m = fmaxf(m, __shfl_xor(m, 1, 32));
      m = fmaxf(m, __shfl_xor(m, 2, 32));
      m = fmaxf(m, __shfl_xor(m, 4, 32));
      m = fmaxf(m, __shfl_xor(m, 8, 32));
      tm[r] = m;
    }

    // ---- online softmax rescale ----
#pragma unroll
    for (int r = 0; r < 8; ++r) {
      const float mn = fmaxf(row_m[r], tm[r]);
      const float al = __expf(row_m[r] - mn);
      row_m[r] = mn;
      row_l[r] *= al;
#pragma unroll
      for (int t = 0; t < 8; ++t) oacc[t][r] *= al;
    }

    // ---- P = exp(S - m), row sums ----
#pragma unroll
    for (int r = 0; r < 8; ++r) {
      float s = 0.0f;
#pragma unroll
      for (int nt = 0; nt < 4; ++nt) {
        const float p = __expf(sc[nt][r] - row_m[r]);
        sc[nt][r] = p;
        s += p;
      }
      s += __shfl_xor(s, 1, 32);
      s += __shfl_xor(s, 2, 32);
      s += __shfl_xor(s, 4, 32);
      s += __shfl_xor(s, 8, 32);
      row_l[r] += s;
    }

    // ---- write P (f16) to wave-private LDS rows, re-read as A-fragments ----
#pragma unroll
    for (int nt = 0; nt < 4; ++nt)
#pragma unroll
      for (int r = 0; r < 8; ++r)
        PL[(w * 16 + r + 8 * hf) * 72 + nt * 16 + nl] = (_Float16)sc[nt][r];

    // ---- O += P * V ----
#pragma unroll
    for (int ks = 0; ks < 2; ++ks) {
      const v16h pf = load_a_frag(&PL[(w * 16) * 72 + ks * 32], 72);
#pragma unroll
      for (int t = 0; t < 8; ++t) {
        const v16h vf = load_b_frag(&VL[(ks * 32) * 136 + t * 16], 136);
        oacc[t] = wmma16(pf, vf, oacc[t]);
      }
    }
  }

  // ---- normalize and store context (f16, heads re-interleaved) ----
#pragma unroll
  for (int t = 0; t < 8; ++t)
#pragma unroll
    for (int r = 0; r < 8; ++r) {
      const int mg = rowbase + w * 16 + r + 8 * hf;
      Cg[mg * H_ + colbase + t * 16 + nl] = (_Float16)(oacc[t][r] / row_l[r]);
    }
}

// ---------------------------------------------------------------------------
extern "C" void kernel_launch(void* const* d_in, const int* in_sizes, int n_in,
                              void* d_out, int out_size, void* d_ws, size_t ws_size,
                              hipStream_t stream) {
  const float* x  = (const float*)d_in[0];
  // d_in[1] = additive causal mask (applied analytically in attn128)
  const float* wq = (const float*)d_in[2];
  const float* bq = (const float*)d_in[3];
  const float* wk = (const float*)d_in[4];
  const float* bk = (const float*)d_in[5];
  const float* wv = (const float*)d_in[6];
  const float* bv = (const float*)d_in[7];
  const float* wo = (const float*)d_in[8];
  const float* bo = (const float*)d_in[9];

  const int NX = B_ * S_ * H_;  // 8388608
  const int NW = H_ * H_;       // 4194304

  _Float16* xh  = (_Float16*)d_ws;
  _Float16* wqh = xh + NX;
  _Float16* wkh = wqh + NW;
  _Float16* wvh = wkh + NW;
  _Float16* woh = wvh + NW;
  _Float16* qh  = woh + NW;
  _Float16* kh  = qh + NX;
  _Float16* vh  = kh + NX;
  _Float16* ch  = vh + NX;   // total ~117 MB of workspace

  cast_f16_kernel<<<1024, 256, 0, stream>>>(x,  xh,  NX);
  cast_f16_kernel<<<1024, 256, 0, stream>>>(wq, wqh, NW);
  cast_f16_kernel<<<1024, 256, 0, stream>>>(wk, wkh, NW);
  cast_f16_kernel<<<1024, 256, 0, stream>>>(wv, wvh, NW);
  cast_f16_kernel<<<1024, 256, 0, stream>>>(wo, woh, NW);

  const dim3 gGemm(H_ / 128, (B_ * S_) / 128);  // 16 x 32
  gemm128<true><<<gGemm, 256, 0, stream>>>(xh, wqh, bq, qh, B_ * S_, H_, H_);
  gemm128<true><<<gGemm, 256, 0, stream>>>(xh, wkh, bk, kh, B_ * S_, H_, H_);
  gemm128<true><<<gGemm, 256, 0, stream>>>(xh, wvh, bv, vh, B_ * S_, H_, H_);

  attn128<<<dim3(S_ / 128, NH_, B_), 256, 0, stream>>>(qh, kh, vh, ch);

  gemm128<false><<<gGemm, 256, 0, stream>>>(ch, woh, bo, d_out, B_ * S_, H_, H_);
}
